// SpatialAttentionFuser_85478439125170
// MI455X (gfx1250) — compile-verified, matrix operand order checked by reading, not run on hardware
//
#include <hip/hip_runtime.h>
#include <cstdint>
#include <cstddef>

// ---------------------------------------------------------------------------
// MI455X (gfx1250, wave32) implementation of SpatialAttentionFuser forward.
// - All GEMM-shaped math on v_wmma_f32_16x16x32_bf16 (bf16 in, f32 acc).
// - Global->LDS tile staging via the Tensor Data Mover (tensor_load_to_lds),
//   double-buffered in the GEMM so DMA overlaps the WMMA pipe.
// - GEMM waves own 64x64 tiles (4x4 WMMA accs): 16 WMMA per 8 LDS fragment
//   loads -> 2x the LDS-byte reuse of a 32x64 tile.
// - Density bias is constant along the softmax axis -> dropped (invariant).
// ---------------------------------------------------------------------------

typedef __bf16 bf16_t;
typedef unsigned int u32;
typedef unsigned long long u64;
typedef __attribute__((ext_vector_type(16))) __bf16 v16bf;
typedef __attribute__((ext_vector_type(8)))  float  v8f;
typedef __attribute__((ext_vector_type(4)))  u32    su32x4;
typedef __attribute__((ext_vector_type(8)))  u32    su32x8;

#define NBATCH  2
#define LSEQ    1024
#define DMODEL  768
#define NHEADS  12
#define HDIM    64
#define NLAYERS 6
#define DMLP    3072

union V8 { v8f v; float f[8]; };

// ---------------------------------------------------------------------------
// TDM: load a 2D tile (tile_rows x tile_cols bf16 elements) from a row-major
// tensor (row stride in elements) into LDS at lds_byte_off.  pad_fields adds
// optional LDS padding (D# group1 dword0 bits 20..31).  Wave-level DMA,
// tracked with TENSORcnt.  D# layout per CDNA5 ISA §8.3/8.4.
// ---------------------------------------------------------------------------
__device__ __forceinline__ void tdm_load_2d(u32 lds_byte_off, const void* gptr,
                                            u32 tile_cols, u32 tile_rows,
                                            u32 row_stride_elems, u32 pad_fields)
{
  u64 ga = (u64)(uintptr_t)gptr;
  su32x4 g0;
  g0[0] = 1u;                                            // count=1, user load
  g0[1] = lds_byte_off;                                  // lds_addr
  g0[2] = (u32)ga;                                       // global_addr[31:0]
  g0[3] = ((u32)(ga >> 32) & 0x01FFFFFFu) | (2u << 30);  // addr[56:32] | type=2
  u32 td0 = row_stride_elems;       // tensor_dim0 (OOB bound; X < td0 always)
  u32 td1 = 0x100000u;              // tensor_dim1: large (tiles never OOB)
  su32x8 g1;
  g1[0] = (1u << 16) | pad_fields;  // data_size=2B | pad cfg; wg_mask=0
  g1[1] = (td0 & 0xFFFFu) << 16;                          // td0[15:0]
  g1[2] = ((td0 >> 16) & 0xFFFFu) | ((td1 & 0xFFFFu) << 16);
  g1[3] = ((td1 >> 16) & 0xFFFFu) | (tile_cols << 16);    // tile_dim0
  g1[4] = tile_rows;                                      // tile_dim1 (dim2=0)
  g1[5] = row_stride_elems;                               // dim0_stride[31:0]
  g1[6] = 0u;
  g1[7] = 0u;
  asm volatile("tensor_load_to_lds %0, %1" :: "s"(g0), "s"(g1) : "memory");
}

// pad cfg: +4 DWORDs (8 bf16) after every 16 DWORDs (32 bf16) -> LDS row
// stride 40 elems, conflict-free fragment reads.
#define TDM_PAD_40 ((1u << 20) | (3u << 22) | (3u << 25))

// Load a 16x32 bf16 WMMA fragment (A-layout; B mirrored with "row" = column).
// lane<16: row=lane, K=(0..7,16..23); lane>=16: row=lane-16, K=(8..15,24..31).
__device__ __forceinline__ v16bf ldfrag(const bf16_t* __restrict__ base, int stride) {
  const int lane = threadIdx.x & 31;
  const bf16_t* p = base + (lane & 15) * stride + ((lane >> 4) << 3);
  union { uint4 u[2]; v16bf v; } t;
  t.u[0] = *(const uint4*)(p);
  t.u[1] = *(const uint4*)(p + 16);
  return t.v;
}

__device__ __forceinline__ v8f wmma_bf16(v16bf a, v16bf b, v8f c) {
  return __builtin_amdgcn_wmma_f32_16x16x32_bf16(false, a, false, b, (short)0, c,
                                                 false, false);
}

// ---------------------------------------------------------------------------
// Tiled GEMM:  out[M,N] = A[M,K](bf16) @ Wt[N,K](bf16, pre-transposed) + epi.
// Block tile 128x128, BK=32, 128 threads = 4 waves in a 2x2 grid; each wave
// computes a 64x64 tile = 4x4 WMMA accumulators (16 WMMA / k-step / wave).
// A/B tiles DMA'd by the TDM (wave0/wave1), double-buffered.
// ---------------------------------------------------------------------------
#define GBM 128
#define GBN 128
#define GBK 32
#define GPAD 40
#define GTILE (GBM * GPAD)

__global__ __launch_bounds__(128) void gemm_bf16_kernel(
    const bf16_t* __restrict__ A, const bf16_t* __restrict__ Wt,
    const float* __restrict__ bias, const float* __restrict__ resid,
    const float* __restrict__ bcast, int bdiv,
    float* __restrict__ outF, bf16_t* __restrict__ outB,
    int M, int N, int K, int doGelu)
{
  __shared__ bf16_t As[2][GTILE];
  __shared__ bf16_t Bs[2][GTILE];
  const int tid  = threadIdx.x;
  const int lane = tid & 31;
  const int wave = tid >> 5;
  const int uwave = __builtin_amdgcn_readfirstlane(wave);  // uniform branch
  const int wm   = wave & 1;        // 2 waves along M (64 rows each)
  const int wn   = wave >> 1;       // 2 waves along N (64 cols each)
  const int bm   = blockIdx.x * GBM;
  const int bn   = blockIdx.y * GBN;

  const bf16_t* Aptr = A  + (size_t)bm * K;   // tile row block, advance along K
  const bf16_t* Bptr = Wt + (size_t)bn * K;
  const u32 asOff = (u32)(uintptr_t)&As[0][0];
  const u32 bsOff = (u32)(uintptr_t)&Bs[0][0];

  // Prologue: DMA tile 0.
  if (uwave == 0)      tdm_load_2d(asOff, Aptr, GBK, GBM, K, TDM_PAD_40);
  else if (uwave == 1) tdm_load_2d(bsOff, Bptr, GBK, GBM, K, TDM_PAD_40);

  v8f acc[4][4] = {};
  const int nk = K / GBK;
  for (int t = 0; t < nk; ++t) {
    const int sel = t & 1;
    if (uwave <= 1) __builtin_amdgcn_s_wait_tensorcnt(0);
    __syncthreads();   // tile t ready; tile t^1 fully consumed (prev iter)
    if (t + 1 < nk) {
      const int ns = sel ^ 1;
      if (uwave == 0)
        tdm_load_2d(asOff + ns * (GTILE * 2), Aptr + (size_t)(t + 1) * GBK,
                    GBK, GBM, K, TDM_PAD_40);
      else if (uwave == 1)
        tdm_load_2d(bsOff + ns * (GTILE * 2), Bptr + (size_t)(t + 1) * GBK,
                    GBK, GBM, K, TDM_PAD_40);
    }
    v16bf bf[4];
#pragma unroll
    for (int ni = 0; ni < 4; ++ni)
      bf[ni] = ldfrag(&Bs[sel][(wn * 64 + ni * 16) * GPAD], GPAD);
#pragma unroll
    for (int mi = 0; mi < 4; ++mi) {
      v16bf af = ldfrag(&As[sel][(wm * 64 + mi * 16) * GPAD], GPAD);
#pragma unroll
      for (int ni = 0; ni < 4; ++ni)
        acc[mi][ni] = wmma_bf16(af, bf[ni], acc[mi][ni]);
    }
  }

  // Epilogue. C/D layout: vgpr e -> row e (lanes 0-15) / e+8 (lanes 16-31).
#pragma unroll
  for (int mi = 0; mi < 4; ++mi) {
#pragma unroll
    for (int ni = 0; ni < 4; ++ni) {
      V8 u; u.v = acc[mi][ni];
#pragma unroll
      for (int e = 0; e < 8; ++e) {
        int m = bm + wm * 64 + mi * 16 + e + ((lane >> 4) << 3);
        int n = bn + wn * 64 + ni * 16 + (lane & 15);
        float val = u.f[e];
        if (bias)  val += bias[n];
        if (bcast) val += bcast[(m / bdiv) * N + n];
        if (resid) val += resid[(size_t)m * N + n];
        if (doGelu) val = 0.5f * val * (1.0f + erff(val * 0.70710678118654752f));
        if (outF) outF[(size_t)m * N + n] = val;
        if (outB) outB[(size_t)m * N + n] = (bf16_t)val;
      }
    }
  }
}

// ---------------------------------------------------------------------------
// Flash attention: one block per (b, h, 128-q-rows); 8 waves, 16 q rows each.
// K / V^T tiles DMA'd by TDM; P staged in LDS (reusing the K region).
// ---------------------------------------------------------------------------
__global__ __launch_bounds__(256) void attn_kernel(
    const bf16_t* __restrict__ Q,   // (B,NH,L,HD)
    const bf16_t* __restrict__ Km,  // (B,NH,L,HD)
    const bf16_t* __restrict__ Vt,  // (B,NH,HD,L)
    bf16_t* __restrict__ O)         // (B,L,D)
{
  __shared__ bf16_t smA[16384];  // union: K tile [128][64]  /  P [8][16][128]
  __shared__ bf16_t smV[8192];   // Vt tile [64][128]
  const int tid = threadIdx.x, lane = tid & 31, wave = tid >> 5;
  const int uwave = __builtin_amdgcn_readfirstlane(wave);
  const int qt = blockIdx.x, h = blockIdx.y, b = blockIdx.z;
  const size_t headoff = ((size_t)(b * NHEADS + h)) * LSEQ * HDIM;
  const u32 smAoff = (u32)(uintptr_t)&smA[0];
  const u32 smVoff = (u32)(uintptr_t)&smV[0];

  const bf16_t* qbase = Q + headoff + (size_t)(qt * 128 + wave * 16) * HDIM;
  v16bf qf0 = ldfrag(qbase, HDIM);        // d = 0..31
  v16bf qf1 = ldfrag(qbase + 32, HDIM);   // d = 32..63

  float runM[8], runS[8];
  v8f oacc[4] = {};
#pragma unroll
  for (int v = 0; v < 8; ++v) { runM[v] = -3.0e38f; runS[v] = 0.0f; }
  const float scl = 0.125f;   // HD^-0.5

  for (int kt = 0; kt < LSEQ / 128; ++kt) {
    __syncthreads();   // all P/V reads of previous tile complete
    if (uwave == 0) {        // K tile: 128x64 contiguous = one 8192-elem line
      tdm_load_2d(smAoff, Km + headoff + (size_t)(kt * 128) * HDIM,
                  8192, 1, 8192, 0);
      __builtin_amdgcn_s_wait_tensorcnt(0);
    } else if (uwave == 1) { // V^T tile: 64 rows x 128 keys, row stride L
      tdm_load_2d(smVoff, Vt + headoff + kt * 128, 128, 64, LSEQ, 0);
      __builtin_amdgcn_s_wait_tensorcnt(0);
    }
    __syncthreads();

    // S = Q K^T : 8 key-tiles of 16, contraction d=64 in two WMMA steps.
    v8f sacc[8] = {};
#pragma unroll
    for (int j = 0; j < 8; ++j) {
      sacc[j] = wmma_bf16(qf0, ldfrag(&smA[(j * 16) * HDIM], HDIM), sacc[j]);
      sacc[j] = wmma_bf16(qf1, ldfrag(&smA[(j * 16) * HDIM + 32], HDIM), sacc[j]);
    }
    __syncthreads();   // K reads done; smA region reusable for P

    V8 su[8];
#pragma unroll
    for (int j = 0; j < 8; ++j) su[j].v = sacc[j];

    float pm[8];
#pragma unroll
    for (int v = 0; v < 8; ++v) {
      float m = su[0].f[v];
#pragma unroll
      for (int j = 1; j < 8; ++j) m = fmaxf(m, su[j].f[v]);
      pm[v] = m;
    }
#pragma unroll
    for (int mask = 1; mask < 16; mask <<= 1)
#pragma unroll
      for (int v = 0; v < 8; ++v) pm[v] = fmaxf(pm[v], __shfl_xor(pm[v], mask, 32));

    float corr[8], ps[8];
#pragma unroll
    for (int v = 0; v < 8; ++v) {
      float nm = fmaxf(runM[v], pm[v] * scl);
      corr[v] = __expf(runM[v] - nm);
      runM[v] = nm;
      ps[v] = 0.0f;
    }

    bf16_t* Pw = &smA[wave * 16 * 128];
#pragma unroll
    for (int j = 0; j < 8; ++j) {
#pragma unroll
      for (int v = 0; v < 8; ++v) {
        float p = __expf(su[j].f[v] * scl - runM[v]);
        ps[v] += p;
        int row = v + ((lane >> 4) << 3);
        int col = j * 16 + (lane & 15);
        Pw[row * 128 + col] = (bf16_t)p;
      }
    }
#pragma unroll
    for (int mask = 1; mask < 16; mask <<= 1)
#pragma unroll
      for (int v = 0; v < 8; ++v) ps[v] += __shfl_xor(ps[v], mask, 32);
#pragma unroll
    for (int v = 0; v < 8; ++v) runS[v] = runS[v] * corr[v] + ps[v];

#pragma unroll
    for (int d = 0; d < 4; ++d) {
      V8 u; u.v = oacc[d];
#pragma unroll
      for (int v = 0; v < 8; ++v) u.f[v] *= corr[v];
      oacc[d] = u.v;
    }
    __syncthreads();   // P visible

    // O += P V : contraction over 128 keys = 4 WMMA k-steps, 4 d-tiles.
#pragma unroll
    for (int t = 0; t < 4; ++t) {
      v16bf pf = ldfrag(Pw + t * 32, 128);
#pragma unroll
      for (int d = 0; d < 4; ++d) {
        v16bf vf = ldfrag(&smV[(d * 16) * 128 + t * 32], 128);
        oacc[d] = wmma_bf16(pf, vf, oacc[d]);
      }
    }
  }

#pragma unroll
  for (int d = 0; d < 4; ++d) {
    V8 u; u.v = oacc[d];
#pragma unroll
    for (int v = 0; v < 8; ++v) {
      int m = qt * 128 + wave * 16 + v + ((lane >> 4) << 3);
      int n = h * HDIM + d * 16 + (lane & 15);
      O[((size_t)(b * LSEQ + m)) * DMODEL + n] = (bf16_t)(u.f[v] / runS[v]);
    }
  }
}

// ---------------------------------------------------------------------------
// RMSNorm + RoPE on q,k; convert+transpose for v.
// ---------------------------------------------------------------------------
__global__ __launch_bounds__(256) void rope_kernel(
    const float* __restrict__ QKV,   // (B,L,3,NH,HD)
    const float* __restrict__ qsc, const float* __restrict__ ksc,
    const float* __restrict__ cosT, const float* __restrict__ sinT,  // (L,32)
    bf16_t* __restrict__ Qo, bf16_t* __restrict__ Ko,                // (B,NH,L,HD)
    bf16_t* __restrict__ Vto)                                        // (B,NH,HD,L)
{
  const int row = blockIdx.x;               // b*L + l
  const int b = row / LSEQ, l = row % LSEQ;
  const int tid = threadIdx.x, lane = tid & 31, wave = tid >> 5;
  const float* base = QKV + (size_t)row * (3 * DMODEL);

  for (int task = wave; task < 2 * NHEADS; task += 8) {
    int hh = task >> 1, isK = task & 1;
    const float* src = base + isK * DMODEL + hh * HDIM;
    const float* sc  = isK ? ksc : qsc;
    float x0 = src[2 * lane], x1 = src[2 * lane + 1];
    float ss = x0 * x0 + x1 * x1;
#pragma unroll
    for (int mask = 1; mask < 32; mask <<= 1) ss += __shfl_xor(ss, mask, 32);
    float inv = rsqrtf(ss * (1.0f / 64.0f) + 1e-6f);
    float y0 = x0 * inv * sc[2 * lane];
    float y1 = x1 * inv * sc[2 * lane + 1];
    float c = cosT[l * 32 + lane], s = sinT[l * 32 + lane];
    float o0 = c * y0 - s * y1;
    float o1 = s * y0 + c * y1;
    bf16_t* dst = (isK ? Ko : Qo) + ((size_t)(b * NHEADS + hh) * LSEQ + l) * HDIM;
    dst[2 * lane]     = (bf16_t)o0;
    dst[2 * lane + 1] = (bf16_t)o1;
  }
  for (int idx = tid; idx < DMODEL; idx += 256) {
    int hh = idx >> 6, d = idx & 63;
    Vto[((size_t)(b * NHEADS + hh) * HDIM + d) * LSEQ + l] =
        (bf16_t)base[2 * DMODEL + idx];
  }
}

// ---------------------------------------------------------------------------
// LayerNorm: one block per row of 768; outputs bf16.
// ---------------------------------------------------------------------------
__global__ __launch_bounds__(256) void ln_kernel(
    const float* __restrict__ X, const float* __restrict__ g,
    const float* __restrict__ bb, bf16_t* __restrict__ H)
{
  __shared__ float red[16];
  const int row = blockIdx.x, tid = threadIdx.x, lane = tid & 31, wave = tid >> 5;
  const float* x = X + (size_t)row * DMODEL;
  float v0 = x[tid], v1 = x[tid + 256], v2 = x[tid + 512];
  float s = v0 + v1 + v2;
  float q = v0 * v0 + v1 * v1 + v2 * v2;
#pragma unroll
  for (int mask = 1; mask < 32; mask <<= 1) {
    s += __shfl_xor(s, mask, 32);
    q += __shfl_xor(q, mask, 32);
  }
  if (lane == 0) { red[wave] = s; red[8 + wave] = q; }
  __syncthreads();
  float ts = 0.0f, tq = 0.0f;
#pragma unroll
  for (int w = 0; w < 8; ++w) { ts += red[w]; tq += red[8 + w]; }
  float mean = ts * (1.0f / 768.0f);
  float var  = tq * (1.0f / 768.0f) - mean * mean;
  float inv  = rsqrtf(var + 1e-5f);
  bf16_t* hrow = H + (size_t)row * DMODEL;
  hrow[tid]       = (bf16_t)((v0 - mean) * inv * g[tid]       + bb[tid]);
  hrow[tid + 256] = (bf16_t)((v1 - mean) * inv * g[tid + 256] + bb[tid + 256]);
  hrow[tid + 512] = (bf16_t)((v2 - mean) * inv * g[tid + 512] + bb[tid + 512]);
}

// fp32 -> bf16 convert (activations)
__global__ void cvt_kernel(const float* __restrict__ in, bf16_t* __restrict__ out,
                           int n) {
  int i = blockIdx.x * 256 + threadIdx.x;
  if (i < n) out[i] = (bf16_t)in[i];
}

// Weight prep: in [K,N] f32 (layer blkIdx.z) -> out [N,K] bf16 (transposed),
// LDS-tiled so both global accesses are coalesced.
__global__ __launch_bounds__(256) void transpose_cvt_kernel(
    const float* __restrict__ in, bf16_t* __restrict__ out, int K, int N)
{
  __shared__ float tile[32][33];
  const size_t moff = (size_t)blockIdx.z * K * N;
  const float* src = in + moff;
  bf16_t* dst = out + moff;
  const int bn = blockIdx.x * 32;
  const int bk = blockIdx.y * 32;
  const int tx = threadIdx.x & 31, ty = threadIdx.x >> 5;
#pragma unroll
  for (int r = 0; r < 32; r += 8)
    tile[ty + r][tx] = src[(size_t)(bk + ty + r) * N + bn + tx];
  __syncthreads();
#pragma unroll
  for (int r = 0; r < 32; r += 8)
    dst[(size_t)(bn + ty + r) * K + bk + tx] = (bf16_t)tile[tx][ty + r];
}

// RoPE tables (L,32): pairs 0..15 use x=l%32, 16..31 use y=l/32.
__global__ void pe_kernel(float* __restrict__ cosT, float* __restrict__ sinT) {
  int idx = blockIdx.x * 256 + threadIdx.x;
  if (idx >= LSEQ * 32) return;
  int l = idx >> 5, p = idx & 31;
  int j = (p < 16) ? p : p - 16;
  float pos = (p < 16) ? (float)(l & 31) : (float)(l >> 5);
  float omega = powf(10000.0f, -(float)j / 16.0f);
  float a = pos * omega;
  cosT[idx] = cosf(a);
  sinT[idx] = sinf(a);
}

// condp[b,n] = cond[b,:] @ cond_w[:,n] + cond_b[n]   (tiny)
__global__ __launch_bounds__(256) void cond_kernel(
    const float* __restrict__ cond, const float* __restrict__ W,
    const float* __restrict__ bias, float* __restrict__ out)
{
  int idx = blockIdx.x * 256 + threadIdx.x;
  if (idx >= NBATCH * DMODEL) return;
  int b = idx / DMODEL, n = idx % DMODEL;
  float s = bias[n];
  for (int k = 0; k < 1536; ++k) s += cond[b * 1536 + k] * W[(size_t)k * DMODEL + n];
  out[idx] = s;
}

// ---------------------------------------------------------------------------
extern "C" void kernel_launch(void* const* d_in, const int* in_sizes, int n_in,
                              void* d_out, int out_size, void* d_ws, size_t ws_size,
                              hipStream_t stream)
{
  (void)in_sizes; (void)n_in; (void)out_size; (void)ws_size;
  const float* spatial   = (const float*)d_in[0];
  // d_in[1] density_weights, d_in[14] dscale, d_in[15] dbias: bias is constant
  // along the softmax axis -> mathematically a no-op; intentionally unused.
  const float* cond      = (const float*)d_in[2];
  const float* spatial_w = (const float*)d_in[3];
  const float* spatial_b = (const float*)d_in[4];
  const float* cond_w    = (const float*)d_in[5];
  const float* cond_b    = (const float*)d_in[6];
  const float* ln1_g     = (const float*)d_in[7];
  const float* ln1_b     = (const float*)d_in[8];
  const float* qkv_w     = (const float*)d_in[9];
  const float* q_scale   = (const float*)d_in[10];
  const float* k_scale   = (const float*)d_in[11];
  const float* proj_w    = (const float*)d_in[12];
  const float* proj_b    = (const float*)d_in[13];
  const float* ln2_g     = (const float*)d_in[16];
  const float* ln2_b     = (const float*)d_in[17];
  const float* mlp_w1    = (const float*)d_in[18];
  const float* mlp_b1    = (const float*)d_in[19];
  const float* mlp_w2    = (const float*)d_in[20];
  const float* mlp_b2    = (const float*)d_in[21];
  const float* out_w     = (const float*)d_in[22];
  const float* out_b     = (const float*)d_in[23];

  char* ws = (char*)d_ws;
  size_t off = 0;
  auto alloc = [&](size_t bytes) -> void* {
    void* p = ws + off;
    off = (off + bytes + 255) & ~(size_t)255;
    return p;
  };

  // bf16 weights, stored transposed: Wt[N][K]
  bf16_t* wqkv  = (bf16_t*)alloc((size_t)NLAYERS * DMODEL * 3 * DMODEL * 2);
  bf16_t* wproj = (bf16_t*)alloc((size_t)NLAYERS * DMODEL * DMODEL * 2);
  bf16_t* wm1   = (bf16_t*)alloc((size_t)NLAYERS * DMODEL * DMLP * 2);
  bf16_t* wm2   = (bf16_t*)alloc((size_t)NLAYERS * DMLP * DMODEL * 2);
  bf16_t* wsp   = (bf16_t*)alloc((size_t)DMODEL * DMODEL * 2);
  bf16_t* wout  = (bf16_t*)alloc((size_t)DMODEL * DMODEL * 2);
  float*  cosT  = (float*)alloc((size_t)LSEQ * 32 * 4);
  float*  sinT  = (float*)alloc((size_t)LSEQ * 32 * 4);
  float*  condp = (float*)alloc((size_t)NBATCH * DMODEL * 4);
  float*  X     = (float*)alloc((size_t)NBATCH * LSEQ * DMODEL * 4);
  bf16_t* hb    = (bf16_t*)alloc((size_t)NBATCH * LSEQ * DMODEL * 2);
  float*  QKVb  = (float*)alloc((size_t)NBATCH * LSEQ * 3 * DMODEL * 4);
  bf16_t* Qb    = (bf16_t*)alloc((size_t)NBATCH * NHEADS * LSEQ * HDIM * 2);
  bf16_t* Kb    = (bf16_t*)alloc((size_t)NBATCH * NHEADS * LSEQ * HDIM * 2);
  bf16_t* Vtb   = (bf16_t*)alloc((size_t)NBATCH * NHEADS * HDIM * LSEQ * 2);
  bf16_t* AOb   = (bf16_t*)alloc((size_t)NBATCH * LSEQ * DMODEL * 2);
  bf16_t* MIDb  = (bf16_t*)alloc((size_t)NBATCH * LSEQ * DMLP * 2);

  const int M = NBATCH * LSEQ;   // 2048

  auto cvt = [&](const float* src, bf16_t* dst, size_t n) {
    cvt_kernel<<<(unsigned)((n + 255) / 256), 256, 0, stream>>>(src, dst, (int)n);
  };
  auto tcvt = [&](const float* src, bf16_t* dst, int Kd, int Nd, int nl) {
    dim3 g(Nd / 32, Kd / 32, nl);
    transpose_cvt_kernel<<<g, 256, 0, stream>>>(src, dst, Kd, Nd);
  };
  auto gemm = [&](const bf16_t* Aa, const bf16_t* Wtt, const float* bias,
                  const float* resid, const float* bcast, int bdiv,
                  float* oF, bf16_t* oB, int Mm, int Nn, int Kk, int gelu) {
    dim3 g(Mm / GBM, Nn / GBN);
    gemm_bf16_kernel<<<g, 128, 0, stream>>>(Aa, Wtt, bias, resid, bcast, bdiv,
                                            oF, oB, Mm, Nn, Kk, gelu);
  };

  // Per-call weight prep (transpose + bf16) and activation convert.
  tcvt(qkv_w,  wqkv,  DMODEL, 3 * DMODEL, NLAYERS);
  tcvt(proj_w, wproj, DMODEL, DMODEL,     NLAYERS);
  tcvt(mlp_w1, wm1,   DMODEL, DMLP,       NLAYERS);
  tcvt(mlp_w2, wm2,   DMLP,   DMODEL,     NLAYERS);
  tcvt(spatial_w, wsp, DMODEL, DMODEL, 1);
  tcvt(out_w,  wout,  DMODEL, DMODEL, 1);
  cvt(spatial, hb, (size_t)M * DMODEL);

  pe_kernel<<<(LSEQ * 32 + 255) / 256, 256, 0, stream>>>(cosT, sinT);
  cond_kernel<<<(NBATCH * DMODEL + 255) / 256, 256, 0, stream>>>(cond, cond_w,
                                                                 cond_b, condp);
  // x = spatial @ spatial_w + spatial_b + condp[b]
  gemm(hb, wsp, spatial_b, nullptr, condp, LSEQ, X, nullptr, M, DMODEL, DMODEL, 0);

  for (int i = 0; i < NLAYERS; ++i) {
    ln_kernel<<<M, 256, 0, stream>>>(X, ln1_g + i * DMODEL, ln1_b + i * DMODEL, hb);
    gemm(hb, wqkv + (size_t)i * DMODEL * 3 * DMODEL, nullptr, nullptr, nullptr, 1,
         QKVb, nullptr, M, 3 * DMODEL, DMODEL, 0);
    rope_kernel<<<M, 256, 0, stream>>>(QKVb, q_scale + i * HDIM, k_scale + i * HDIM,
                                       cosT, sinT, Qb, Kb, Vtb);
    attn_kernel<<<dim3(LSEQ / 128, NHEADS, NBATCH), 256, 0, stream>>>(Qb, Kb, Vtb, AOb);
    gemm(AOb, wproj + (size_t)i * DMODEL * DMODEL, proj_b + i * DMODEL, X, nullptr, 1,
         X, nullptr, M, DMODEL, DMODEL, 0);
    ln_kernel<<<M, 256, 0, stream>>>(X, ln2_g + i * DMODEL, ln2_b + i * DMODEL, hb);
    gemm(hb, wm1 + (size_t)i * DMODEL * DMLP, mlp_b1 + i * DMLP, nullptr, nullptr, 1,
         nullptr, MIDb, M, DMLP, DMODEL, 1 /*gelu*/);
    gemm(MIDb, wm2 + (size_t)i * DMLP * DMODEL, mlp_b2 + i * DMODEL, X, nullptr, 1,
         X, nullptr, M, DMODEL, DMLP, 0);
  }

  cvt(X, hb, (size_t)M * DMODEL);
  gemm(hb, wout, out_b, nullptr, nullptr, 1, (float*)d_out, nullptr,
       M, DMODEL, DMODEL, 0);
}
